// AttnBlock_41738492182990
// MI455X (gfx1250) — matrix operand order repaired and checked by reading, not compiled, not run on hardware
//
#include <hip/hip_runtime.h>
#include <hip/hip_bf16.h>
#include <cstdint>
#include <cstddef>

typedef __bf16 bf16;
typedef __attribute__((ext_vector_type(8)))  __bf16 v8bf;
typedef __attribute__((ext_vector_type(16))) __bf16 v16bf;
typedef __attribute__((ext_vector_type(8)))  float  v8f;

#define C_DIM 512
#define NTOK  4096
#define NBAT  8
#define QK_SCALE 0.04419417382415922f   // 512^-0.5, folded into q

// ---------- helpers -------------------------------------------------------

__device__ __forceinline__ v8f zero8() {
  v8f z;
#pragma unroll
  for (int i = 0; i < 8; ++i) z[i] = 0.0f;
  return z;
}

// A fragment (16x32 bf16, M rows x K): lanes 0-15 row M=lane, K chunks {0-7,16-23};
// lanes 16-31 row M=lane-16, K chunks {8-15,24-31}.  Source row-major [M][K].
__device__ __forceinline__ v16bf load_frag_a(const bf16* base, int stride, int lane) {
  const int rr = lane & 15;
  const int hh = lane >> 4;
  const bf16* p = base + (size_t)rr * stride + (hh << 3);
  v8bf lo = *(const v8bf*)(p);
  v8bf hi = *(const v8bf*)(p + 16);
  return __builtin_shufflevector(lo, hi, 0,1,2,3,4,5,6,7,8,9,10,11,12,13,14,15);
}

// B fragment (32x16 bf16, K x N cols): lanes 0-15 col N=lane, K chunks {0-7,16-23};
// lanes 16-31 col N=lane-16, K chunks {8-15,24-31}.  Source is B^T row-major [N][K].
__device__ __forceinline__ v16bf load_frag_b(const bf16* base, int stride, int lane) {
  const int rr = lane & 15;
  const int hh = lane >> 4;
  const bf16* p = base + (size_t)rr * stride + (hh << 3);
  v8bf lo = *(const v8bf*)(p);
  v8bf hi = *(const v8bf*)(p + 16);
  return __builtin_shufflevector(lo, hi, 0,1,2,3,4,5,6,7,8,9,10,11,12,13,14,15);
}

__device__ __forceinline__ v8f wmma_bf16(v16bf a, v16bf b, v8f c) {
  return __builtin_amdgcn_wmma_f32_16x16x32_bf16(false, a, false, b, (short)0, c,
                                                 false, false);
}

// ---------- kernel 0: convert fp32 weights -> bf16 ------------------------

__global__ void __launch_bounds__(256) convert_weights_kernel(
    const float* __restrict__ wq, const float* __restrict__ wk,
    const float* __restrict__ wv, const float* __restrict__ wp,
    bf16* __restrict__ wbf) {
  const float* srcs[4] = {wq, wk, wv, wp};
  const int idx = blockIdx.x * 256 + threadIdx.x;      // [0, 262144)
#pragma unroll
  for (int m = 0; m < 4; ++m)
    wbf[(size_t)m * (C_DIM * C_DIM) + idx] = (bf16)srcs[m][idx];
}

// ---------- kernel 1: GroupNorm, write h as [B, N, C] bf16 ----------------

__global__ void __launch_bounds__(256) groupnorm_kernel(
    const float* __restrict__ x, const float* __restrict__ gamma,
    const float* __restrict__ beta, bf16* __restrict__ hbf) {
  __shared__ float s_sum[256];
  __shared__ float s_sq[256];
  const int g = blockIdx.x;              // 32 groups of 16 channels
  const int b = blockIdx.y;
  const int tid = threadIdx.x;
  const float4* xg4 = (const float4*)(x + ((size_t)b * C_DIM + g * 16) * NTOK);

  float s = 0.f, ss = 0.f;
  for (int i = tid; i < 16384; i += 256) {   // 16 ch * 4096 px / 4
    float4 v = xg4[i];
    s  += v.x + v.y + v.z + v.w;
    ss += v.x * v.x + v.y * v.y + v.z * v.z + v.w * v.w;
  }
  s_sum[tid] = s; s_sq[tid] = ss;
  __syncthreads();
  for (int st = 128; st > 0; st >>= 1) {
    if (tid < st) { s_sum[tid] += s_sum[tid + st]; s_sq[tid] += s_sq[tid + st]; }
    __syncthreads();
  }
  const float mu   = s_sum[0] * (1.0f / 65536.0f);
  const float var  = s_sq[0] * (1.0f / 65536.0f) - mu * mu;
  const float rstd = rsqrtf(var + 1e-5f);

  bf16* hb = hbf + (size_t)b * NTOK * C_DIM;
  for (int i = tid; i < 16384; i += 256) {
    float4 v = xg4[i];
    const int e  = i * 4;
    const int c  = g * 16 + (e >> 12);
    const int n  = e & 4095;
    const float ga = gamma[c] * rstd;
    const float be = beta[c] - mu * ga;
    bf16* dst = hb + c;
    dst[(size_t)(n + 0) * C_DIM] = (bf16)(v.x * ga + be);
    dst[(size_t)(n + 1) * C_DIM] = (bf16)(v.y * ga + be);
    dst[(size_t)(n + 2) * C_DIM] = (bf16)(v.z * ga + be);
    dst[(size_t)(n + 3) * C_DIM] = (bf16)(v.w * ga + be);
  }
}

// ---------- kernel 2: QKV projection (q pre-scaled by C^-0.5) -------------
// q/k/v[b][n][o] = sum_c h[b][n][c] * w[o][c] + bias[o]

__global__ void __launch_bounds__(256, 1) qkv_kernel(
    const bf16* __restrict__ hbf, const bf16* __restrict__ wbf,
    const float* __restrict__ bq, const float* __restrict__ bk,
    const float* __restrict__ bv,
    bf16* __restrict__ qb, bf16* __restrict__ kb, bf16* __restrict__ vb) {
  __shared__ bf16 Hs[64 * 520];                     // 64 tokens x 512(+8 pad)
  const int nb   = blockIdx.x * 64;
  const int b    = blockIdx.y;
  const int tid  = threadIdx.x;
  const int lane = tid & 31;
  const int wave = tid >> 5;
  const int rr   = lane & 15;
  const int hh   = lane >> 4;

  const bf16* hsrc = hbf + ((size_t)b * NTOK + nb) * C_DIM;
#pragma unroll
  for (int i = 0; i < 16; ++i) {                    // 64*512/8 = 4096 vecs
    const int v   = tid + 256 * i;
    const int row = v >> 6;
    const int col = (v & 63) * 8;
    *(v8bf*)(Hs + row * 520 + col) = *(const v8bf*)(hsrc + (size_t)row * C_DIM + col);
  }
  __syncthreads();

  const int rb    = (wave & 3) * 16;                // 4 row blocks
  const int obase = (wave >> 2) * 256;              // 2 out-channel halves
  const float* biases[3] = {bq, bk, bv};
  bf16* outs[3] = {qb, kb, vb};

#pragma unroll
  for (int mi = 0; mi < 3; ++mi) {
    const bf16* wmat = wbf + (size_t)mi * C_DIM * C_DIM;
    const float scale = (mi == 0) ? QK_SCALE : 1.0f;
    bf16* outp = outs[mi] + ((size_t)b * NTOK + nb) * C_DIM;
    for (int ot = 0; ot < 16; ++ot) {
      const int oc = obase + ot * 16;
      v8f acc = zero8();
#pragma unroll
      for (int kc = 0; kc < 16; ++kc) {
        v16bf af  = load_frag_a(Hs + rb * 520 + kc * 32, 520, lane);
        v16bf bfr = load_frag_b(wmat + (size_t)oc * C_DIM + kc * 32, C_DIM, lane);
        acc = wmma_bf16(af, bfr, acc);
      }
      const float bias = biases[mi][oc + rr];
#pragma unroll
      for (int j = 0; j < 8; ++j) {
        const int row = rb + j + 8 * hh;
        outp[(size_t)row * C_DIM + oc + rr] = (bf16)((acc[j] + bias) * scale);
      }
    }
  }
}

// ---------- kernel 3: flash attention -------------------------------------
// 32 query rows per workgroup, 64-key tiles, online softmax, O in VGPRs.

__global__ void __launch_bounds__(256, 1) attn_kernel(
    const bf16* __restrict__ qbuf, const bf16* __restrict__ kbuf,
    const bf16* __restrict__ vbuf, bf16* __restrict__ abuf) {
  __shared__ bf16  Qs[32 * 520];                    // 33.3 KB
  __shared__ bf16  Vts[512 * 72];                   // V^T: [chan][key], 73.7 KB
  __shared__ float Ss[32 * 64];                     // score tile, 8 KB
  __shared__ bf16  Ps[32 * 64];                     // softmax probs, 4 KB
  __shared__ float row_m[32], row_l[32], row_sc[32];

  const int qb0  = blockIdx.x * 32;
  const int b    = blockIdx.y;
  const int tid  = threadIdx.x;
  const int lane = tid & 31;
  const int wave = tid >> 5;
  const int rr   = lane & 15;
  const int hh   = lane >> 4;

  const bf16* qsrc  = qbuf + ((size_t)b * NTOK + qb0) * C_DIM;
  const bf16* kbase = kbuf + (size_t)b * NTOK * C_DIM;
  const bf16* vbase = vbuf + (size_t)b * NTOK * C_DIM;

#pragma unroll
  for (int i = 0; i < 8; ++i) {                     // 32*512/8 = 2048 vecs
    const int v   = tid + 256 * i;
    const int row = v >> 6;
    const int col = (v & 63) * 8;
    *(v8bf*)(Qs + row * 520 + col) = *(const v8bf*)(qsrc + (size_t)row * C_DIM + col);
  }
  if (tid < 32) { row_m[tid] = -3.0e38f; row_l[tid] = 0.0f; }

  v8f acc[8];
#pragma unroll
  for (int t = 0; t < 8; ++t) acc[t] = zero8();
  __syncthreads();

  const int srb = (wave & 1) * 16;                  // S: 2x4 grid of 16x16 tiles
  const int scb = (wave >> 1) * 16;
  const int prb = (wave & 1) * 16;                  // PV: 16 rows x 128 chans
  const int pcb = (wave >> 1) * 128;

  for (int kt = 0; kt < 64; ++kt) {
    const int kb0 = kt * 64;

    // stage V^T tile into LDS: Vts[c][key]
#pragma unroll
    for (int i = 0; i < 16; ++i) {
      const int v   = tid + 256 * i;
      const int key = v >> 6;
      const int c0  = (v & 63) * 8;
      v8bf val = *(const v8bf*)(vbase + (size_t)(kb0 + key) * C_DIM + c0);
#pragma unroll
      for (int j = 0; j < 8; ++j) Vts[(c0 + j) * 72 + key] = val[j];
    }
    if (kt + 1 < 64)
      __builtin_prefetch(kbase + (size_t)(kb0 + 64 + scb) * C_DIM, 0, 1);
    __syncthreads();

    // S = Q * K^T : one 16x16 tile per wave, K=512 -> 16 WMMAs
    {
      v8f s = zero8();
      const bf16* krow = kbase + (size_t)(kb0 + scb) * C_DIM;
#pragma unroll
      for (int kc = 0; kc < 16; ++kc) {
        v16bf af  = load_frag_a(Qs + srb * 520 + kc * 32, 520, lane);
        v16bf bfr = load_frag_b(krow + kc * 32, C_DIM, lane);
        s = wmma_bf16(af, bfr, s);
      }
#pragma unroll
      for (int j = 0; j < 8; ++j) Ss[(srb + j + 8 * hh) * 64 + scb + rr] = s[j];
    }
    __syncthreads();

    // online softmax: 8 lanes per query row (all 256 threads active).
    // Rows partition across waves (tid>>3), so row_m/row_l updates by the
    // sub==0 lane never race with readers of the same row.
    {
      const int row = tid >> 3;                     // 0..31
      const int sub = tid & 7;                      // 0..7
      const float* sr = Ss + row * 64 + sub * 8;
      float mloc = sr[0];
#pragma unroll
      for (int j2 = 1; j2 < 8; ++j2) mloc = fmaxf(mloc, sr[j2]);
#pragma unroll
      for (int m = 1; m < 8; m <<= 1)
        mloc = fmaxf(mloc, __shfl_xor(mloc, m, 32));
      const float m_old = row_m[row];
      const float m_new = fmaxf(m_old, mloc);
      float lsum = 0.0f;
      bf16* pr = Ps + row * 64 + sub * 8;
#pragma unroll
      for (int j2 = 0; j2 < 8; ++j2) {
        const float p = __expf(sr[j2] - m_new);
        lsum += p;
        pr[j2] = (bf16)p;
      }
#pragma unroll
      for (int m = 1; m < 8; m <<= 1)
        lsum += __shfl_xor(lsum, m, 32);
      if (sub == 0) {
        const float alpha = __expf(m_old - m_new);
        row_l[row] = row_l[row] * alpha + lsum;
        row_m[row] = m_new;
        row_sc[row] = alpha;
      }
    }
    __syncthreads();

    // rescale O and accumulate P*V
    float al[8];
#pragma unroll
    for (int j = 0; j < 8; ++j) al[j] = row_sc[prb + j + 8 * hh];
    v16bf af0 = load_frag_a(Ps + prb * 64 + 0,  64, lane);
    v16bf af1 = load_frag_a(Ps + prb * 64 + 32, 64, lane);
#pragma unroll
    for (int t = 0; t < 8; ++t) {
#pragma unroll
      for (int j = 0; j < 8; ++j) acc[t][j] = acc[t][j] * al[j];
      const int ct = pcb + t * 16;
      v16bf b0 = load_frag_b(Vts + (size_t)ct * 72 + 0,  72, lane);
      v16bf b1 = load_frag_b(Vts + (size_t)ct * 72 + 32, 72, lane);
      acc[t] = wmma_bf16(af0, b0, acc[t]);
      acc[t] = wmma_bf16(af1, b1, acc[t]);
    }
    __syncthreads();
  }

  if (tid < 32) row_sc[tid] = 1.0f / row_l[tid];
  __syncthreads();
  float linv[8];
#pragma unroll
  for (int j = 0; j < 8; ++j) linv[j] = row_sc[prb + j + 8 * hh];
  bf16* aout = abuf + ((size_t)b * NTOK + qb0) * C_DIM;
#pragma unroll
  for (int t = 0; t < 8; ++t) {
    const int ct = pcb + t * 16;
#pragma unroll
    for (int j = 0; j < 8; ++j) {
      const int row = prb + j + 8 * hh;
      aout[(size_t)row * C_DIM + ct + rr] = (bf16)(acc[t][j] * linv[j]);
    }
  }
}

// ---------- kernel 4: projection + bias + residual (f32 out) --------------
// out[b][o][n] = x[b][o][n] + bp[o] + sum_c wp[o][c] * a[b][n][c]

__global__ void __launch_bounds__(256, 1) proj_kernel(
    const bf16* __restrict__ abuf, const bf16* __restrict__ wpbf,
    const float* __restrict__ bp, const float* __restrict__ x,
    float* __restrict__ out) {
  __shared__ bf16 As[128 * 520];                    // 130 KB a-tile
  const int nb   = blockIdx.x * 128;
  const int ob   = blockIdx.y * 128;
  const int b    = blockIdx.z;
  const int tid  = threadIdx.x;
  const int lane = tid & 31;
  const int wave = tid >> 5;
  const int rr   = lane & 15;
  const int hh   = lane >> 4;

  const bf16* asrc = abuf + ((size_t)b * NTOK + nb) * C_DIM;
#pragma unroll
  for (int i = 0; i < 32; ++i) {                    // 128*512/8 = 8192 vecs
    const int v   = tid + 256 * i;
    const int row = v >> 6;
    const int col = (v & 63) * 8;
    *(v8bf*)(As + row * 520 + col) = *(const v8bf*)(asrc + (size_t)row * C_DIM + col);
  }
  __syncthreads();

  const int o0 = ob + wave * 16;                    // 16 out-chans per wave
  const bf16* wrow = wpbf + (size_t)o0 * C_DIM;
  for (int nt = 0; nt < 8; ++nt) {
    v8f acc = zero8();
#pragma unroll
    for (int kc = 0; kc < 16; ++kc) {
      v16bf af  = load_frag_a(wrow + kc * 32, C_DIM, lane);
      v16bf bfr = load_frag_b(As + (size_t)nt * 16 * 520 + kc * 32, 520, lane);
      acc = wmma_bf16(af, bfr, acc);
    }
#pragma unroll
    for (int j = 0; j < 8; ++j) {
      const int o = o0 + j + 8 * hh;
      const int n = nb + nt * 16 + rr;
      const size_t idx = ((size_t)b * C_DIM + o) * NTOK + n;
      out[idx] = x[idx] + bp[o] + acc[j];
    }
  }
}

// ---------- launcher ------------------------------------------------------

extern "C" void kernel_launch(void* const* d_in, const int* in_sizes, int n_in,
                              void* d_out, int out_size, void* d_ws, size_t ws_size,
                              hipStream_t stream) {
  const float* x     = (const float*)d_in[0];
  const float* gamma = (const float*)d_in[1];
  const float* beta  = (const float*)d_in[2];
  const float* wq    = (const float*)d_in[3];
  const float* bq    = (const float*)d_in[4];
  const float* wk    = (const float*)d_in[5];
  const float* bk    = (const float*)d_in[6];
  const float* wv    = (const float*)d_in[7];
  const float* bv    = (const float*)d_in[8];
  const float* wp    = (const float*)d_in[9];
  const float* bp    = (const float*)d_in[10];
  float* out = (float*)d_out;

  char* ws = (char*)d_ws;
  constexpr size_t WSZ   = (size_t)C_DIM * C_DIM * sizeof(bf16);          // 512 KB
  constexpr size_t TENS  = (size_t)NBAT * NTOK * C_DIM * sizeof(bf16);    // 32 MB
  bf16* wbf = (bf16*)(ws);                         // 4 weight matrices, bf16
  bf16* hbf = (bf16*)(ws + 4 * WSZ);
  bf16* qb  = (bf16*)(ws + 4 * WSZ + 1 * TENS);
  bf16* kb  = (bf16*)(ws + 4 * WSZ + 2 * TENS);
  bf16* vb  = (bf16*)(ws + 4 * WSZ + 3 * TENS);
  bf16* ab  = (bf16*)(ws + 4 * WSZ + 4 * TENS);

  convert_weights_kernel<<<dim3(C_DIM * C_DIM / 256), 256, 0, stream>>>(
      wq, wk, wv, wp, wbf);
  groupnorm_kernel<<<dim3(32, NBAT), 256, 0, stream>>>(x, gamma, beta, hbf);
  qkv_kernel<<<dim3(NTOK / 64, NBAT), 256, 0, stream>>>(
      hbf, wbf, bq, bk, bv, qb, kb, vb);
  attn_kernel<<<dim3(NTOK / 32, NBAT), 256, 0, stream>>>(qb, kb, vb, ab);
  proj_kernel<<<dim3(NTOK / 128, C_DIM / 128, NBAT), 256, 0, stream>>>(
      ab, wbf + 3 * (size_t)C_DIM * C_DIM, bp, x, out);
}
